// EGATlayer_78151224918205
// MI455X (gfx1250) — matrix operand order repaired and checked by reading, not compiled, not run on hardware
//
#include <hip/hip_runtime.h>
#include <stdint.h>

// ---------------------------------------------------------------------------
// EGAT layer for MI455X (gfx1250), wave32, WMMA 16x16x32 f16 (f32 accumulate).
// - Weights staged TRANSPOSED: A and B fragments are contiguous in LDS
//   (2x ds_load_b128 per fragment).
// - 64 rows per block: 4 row-tiles per wave, B fragments hoisted in registers.
// - Block-uniform full-tile fast path: unguarded stores with immediate
//   offsets; ragged tail tile takes the guarded slow path.
// ---------------------------------------------------------------------------

typedef _Float16 v16h __attribute__((ext_vector_type(16)));
typedef float    v8f  __attribute__((ext_vector_type(8)));

union AFrag { v16h v; _Float16 e[16]; };
union CFrag { v8f  v; float    e[8];  };

#define IN_N 128
#define IN_E 64
#define HF   128          // H * F = 4 * 32 output columns
#define NEG_INF_KEY 0x007FFFFFu   // orderable-uint encoding of -inf

__device__ __forceinline__ unsigned f2key(float f) {
  unsigned u = __float_as_uint(f);
  return (u & 0x80000000u) ? ~u : (u | 0x80000000u);
}
__device__ __forceinline__ float key2f(unsigned k) {
  return (k & 0x80000000u) ? __uint_as_float(k & 0x7FFFFFFFu)
                           : __uint_as_float(~k);
}

// --------------------------------------------------------------- K0: repack
// fp32 weights -> fp16 TRANSPOSED in workspace: wp[n*K + k] = W[k*HF + n].
// layout: [0,16384) W_ni^T | [16384,32768) W_nj^T | [32768,49152) W_node^T |
//         [49152,57344) W_fij^T ([128][64])
__global__ void pack_weights(const float* __restrict__ Wni,
                             const float* __restrict__ Wnj,
                             const float* __restrict__ Wnode,
                             const float* __restrict__ Wfij,
                             _Float16* __restrict__ wp) {
  int i = blockIdx.x * 256 + threadIdx.x;
  if (i < 16384) {
    int n = i >> 7, k = i & 127;
    wp[i] = (_Float16)Wni[k * HF + n];
  } else if (i < 32768) {
    int j = i - 16384; int n = j >> 7, k = j & 127;
    wp[i] = (_Float16)Wnj[k * HF + n];
  } else if (i < 49152) {
    int j = i - 32768; int n = j >> 7, k = j & 127;
    wp[i] = (_Float16)Wnode[k * HF + n];
  } else if (i < 57344) {
    int j = i - 49152; int n = j >> 6, k = j & 63;
    wp[i] = (_Float16)Wfij[k * HF + n];
  }
}

// --------------------------------------------------------------- K_init
__global__ void init_buffers(float* __restrict__ h_out,
                             unsigned* __restrict__ mkeys,
                             float* __restrict__ s, int N) {
  int i = blockIdx.x * 256 + threadIdx.x;
  if (i < N * HF) h_out[i] = 0.0f;
  if (i < N * 4) { mkeys[i] = NEG_INF_KEY; s[i] = 0.0f; }
}

// --------------------------------------------------------------- K1: node GEMMs
// grid.x = ceil(N/64) row-blocks, grid.y = 0:f_ni 1:f_nj 2:h.
// 256 thr = 8 waves; wave w owns 16-col tile c0=16w for 4 row tiles.
__global__ __launch_bounds__(256)
void node_gemm(const float* __restrict__ nfeats,
               const _Float16* __restrict__ wpackT,
               const float* __restrict__ b_node,
               float* __restrict__ f_ni, float* __restrict__ f_nj,
               float* __restrict__ hproj, int N) {
  __shared__ _Float16 ldsA[64 * IN_N];     // 64 x 128 activations (16 KB)
  __shared__ _Float16 ldsWT[HF * IN_N];    // W^T [n][k]            (32 KB)

  const int t     = threadIdx.x;
  const int which = blockIdx.y;
  const int r0    = blockIdx.x * 64;
  const bool full = (r0 + 64 <= N);        // block-uniform

  for (int i = t; i < 64 * IN_N; i += 256) {
    int row = i >> 7, k = i & 127;
    int rr = r0 + row; if (rr >= N) rr = N - 1;
    ldsA[i] = (_Float16)nfeats[(size_t)rr * IN_N + k];
  }
  {
    const uint32_t* s32 = (const uint32_t*)(wpackT + (size_t)which * 16384);
    uint32_t* d32 = (uint32_t*)ldsWT;
    for (int i = t; i < 8192; i += 256) d32[i] = s32[i];
  }
  __syncthreads();

  const int w = t >> 5, lane = t & 31;
  const int c0 = w * 16, col = lane & 15, kh = lane >> 4;

  // Hoisted B fragments; contiguous in transposed LDS -> ds_load_b128 pairs.
  AFrag b[4];
  #pragma unroll
  for (int kt = 0; kt < 4; ++kt)
    #pragma unroll
    for (int hh = 0; hh < 16; ++hh)
      b[kt].e[hh] = ldsWT[(c0 + col) * IN_N + kt * 32 + kh * 16 + hh];

  const float badd = (which == 2) ? b_node[c0 + col] : 0.0f;
  float* out = (which == 0) ? f_ni : (which == 1) ? f_nj : hproj;

  #pragma unroll
  for (int rt = 0; rt < 4; ++rt) {
    CFrag acc;
    #pragma unroll
    for (int i = 0; i < 8; ++i) acc.e[i] = 0.0f;
    #pragma unroll
    for (int kt = 0; kt < 4; ++kt) {
      AFrag a;
      #pragma unroll
      for (int hh = 0; hh < 16; ++hh) {
        int ka = kt * 32 + kh * 16 + hh;
        a.e[hh] = ldsA[(rt * 16 + col) * IN_N + ka];
      }
      acc.v = __builtin_amdgcn_wmma_f32_16x16x32_f16(false, a.v, false, b[kt].v,
                                                     (short)0, acc.v, false, false);
    }
    const int rbase = r0 + rt * 16 + (kh ? 8 : 0);       // row of acc.e[0]
    float* p = out + (size_t)rbase * HF + c0 + col;
    if (full) {
      #pragma unroll
      for (int i = 0; i < 8; ++i) p[i * HF] = acc.e[i] + badd;   // imm offsets
    } else {
      #pragma unroll
      for (int i = 0; i < 8; ++i)
        if (rbase + i < N) p[i * HF] = acc.e[i] + badd;
    }
  }
}

// --------------------------------------------------------------- K2: edge GEMM + fuse
// grid.x = ceil(E/64).  f_fij = efeats @ W_fij fused with gather, bias,
// leaky-ReLU, f_out store, logit contraction and segment-max over dst.
__global__ __launch_bounds__(256)
void edge_gemm(const float* __restrict__ efeats,
               const _Float16* __restrict__ wfijT,
               const float* __restrict__ f_ni, const float* __restrict__ f_nj,
               const float* __restrict__ bias, const float* __restrict__ attn,
               const int* __restrict__ srcI, const int* __restrict__ dstI,
               float* __restrict__ f_out, float* __restrict__ elog,
               unsigned* __restrict__ mkeys, int E) {
  __shared__ _Float16 ldsA[64 * IN_E];     // 64 x 64 edge feats (8 KB)
  __shared__ _Float16 ldsWT[HF * IN_E];    // W_fij^T [n][k]     (16 KB)
  __shared__ float    ldsLog[64 * 4];
  __shared__ int      ldsSrc[64], ldsDst[64];

  const int t  = threadIdx.x;
  const int e0 = blockIdx.x * 64;
  const bool full = (e0 + 64 <= E);        // block-uniform

  for (int i = t; i < 64 * IN_E; i += 256) {
    int row = i >> 6, k = i & 63;
    int ee = e0 + row; if (ee >= E) ee = E - 1;
    ldsA[i] = (_Float16)efeats[(size_t)ee * IN_E + k];
  }
  {
    const uint32_t* s32 = (const uint32_t*)wfijT;
    uint32_t* d32 = (uint32_t*)ldsWT;
    for (int i = t; i < 4096; i += 256) d32[i] = s32[i];
  }
  if (t < 64)        { int ee = e0 + t;      if (ee >= E) ee = E - 1; ldsSrc[t] = srcI[ee]; }
  else if (t < 128)  { int ee = e0 + t - 64; if (ee >= E) ee = E - 1; ldsDst[t - 64] = dstI[ee]; }
  ldsLog[t] = 0.0f;                         // 256 entries = 64 edges x 4 heads
  __syncthreads();

  // Prefetch the gathered f_ni/f_nj rows while the WMMAs run
  // (global_prefetch_b8; each row = 512 B = 4 x 128 B lines).
  #pragma unroll
  for (int j = 0; j < 2; ++j) {
    const int idx  = t + j * 256;           // 0..511
    const int eL   = idx & 63;
    const int part = idx >> 6;              // 0..7
    const float* base = (part < 4) ? (f_ni + (size_t)ldsSrc[eL] * HF)
                                   : (f_nj + (size_t)ldsDst[eL] * HF);
    __builtin_prefetch(base + (part & 3) * 32, 0, 1);
  }

  const int w = t >> 5, lane = t & 31;
  const int c0 = w * 16, col = lane & 15, kh = lane >> 4;

  AFrag b[2];
  #pragma unroll
  for (int kt = 0; kt < 2; ++kt)
    #pragma unroll
    for (int hh = 0; hh < 16; ++hh)
      b[kt].e[hh] = ldsWT[(c0 + col) * IN_E + kt * 32 + kh * 16 + hh];

  const int   c    = c0 + col;
  const float bc   = bias[c];
  const float ac   = attn[c];               // attn is [4][32] row-major == flat[c]
  const int   head = c >> 5;

  #pragma unroll
  for (int rt = 0; rt < 4; ++rt) {
    CFrag acc;
    #pragma unroll
    for (int i = 0; i < 8; ++i) acc.e[i] = 0.0f;
    #pragma unroll
    for (int kt = 0; kt < 2; ++kt) {
      AFrag a;
      #pragma unroll
      for (int hh = 0; hh < 16; ++hh) {
        int ka = kt * 32 + kh * 16 + hh;
        a.e[hh] = ldsA[(rt * 16 + col) * IN_E + ka];
      }
      acc.v = __builtin_amdgcn_wmma_f32_16x16x32_f16(false, a.v, false, b[kt].v,
                                                     (short)0, acc.v, false, false);
    }
    const int Mb = rt * 16 + (kh ? 8 : 0);               // row of acc.e[0]
    float* pout = f_out + (size_t)(e0 + Mb) * HF + c;
    #pragma unroll
    for (int i = 0; i < 8; ++i) {
      const int M = Mb + i;
      float val = acc.e[i]
                + f_ni[(size_t)ldsSrc[M] * HF + c]
                + f_nj[(size_t)ldsDst[M] * HF + c] + bc;
      val = (val > 0.0f) ? val : 0.01f * val;             // leaky relu
      if (full) pout[i * HF] = val;                       // imm-offset store
      else if (e0 + M < E) pout[i * HF] = val;
      float p = val * ac;
      #pragma unroll
      for (int off = 8; off >= 1; off >>= 1) p += __shfl_xor(p, off, 16);
      if (col == 0) atomicAdd(&ldsLog[M * 4 + head], p);
    }
  }
  __syncthreads();

  {
    const int M = t >> 2, hd = t & 3;       // 256 threads = 64 edges x 4 heads
    const int e = e0 + M;
    if (e < E) {
      const float v = ldsLog[t];
      elog[(size_t)e * 4 + hd] = v;
      atomicMax(&mkeys[(size_t)ldsDst[M] * 4 + hd], f2key(v));
    }
  }
}

// --------------------------------------------------------------- K3: exp + segment sum
__global__ void softmax_norm(const int* __restrict__ dstI,
                             const unsigned* __restrict__ mkeys,
                             float* __restrict__ elog, float* __restrict__ s,
                             long E) {
  long t = (long)blockIdx.x * 256 + threadIdx.x;
  if (t >= E * 4) return;
  const long e = t >> 2; const int hd = (int)(t & 3);
  const int d = dstI[e];
  const float m = key2f(mkeys[(size_t)d * 4 + hd]);
  const float ex = expf(elog[t] - m);
  elog[t] = ex;                                   // stash unnormalized weight
  atomicAdd(&s[(size_t)d * 4 + hd], ex);
}

// --------------------------------------------------------------- K4: weighted scatter-add
__global__ void aggregate(const int* __restrict__ srcI, const int* __restrict__ dstI,
                          const float* __restrict__ hproj,
                          const float* __restrict__ elog, const float* __restrict__ s,
                          float* __restrict__ h_out, long E) {
  long t = (long)blockIdx.x * 256 + threadIdx.x;
  if (t >= E * 32) return;
  const long e = t >> 5; const int g = (int)(t & 31);
  const int f0 = g * 4, head = g >> 3;
  const int sE = srcI[e], dE = dstI[e];
  const float a = elog[(size_t)e * 4 + head] / s[(size_t)dE * 4 + head];
  const float4 hv = *(const float4*)(hproj + (size_t)sE * HF + f0);
  float* o = h_out + (size_t)dE * HF + f0;
  atomicAdd(o + 0, hv.x * a);
  atomicAdd(o + 1, hv.y * a);
  atomicAdd(o + 2, hv.z * a);
  atomicAdd(o + 3, hv.w * a);
}

// ---------------------------------------------------------------------------
extern "C" void kernel_launch(void* const* d_in, const int* in_sizes, int n_in,
                              void* d_out, int out_size, void* d_ws, size_t ws_size,
                              hipStream_t stream) {
  const float* nfeats = (const float*)d_in[0];
  const float* efeats = (const float*)d_in[1];
  const int*   srcI   = (const int*)  d_in[2];
  const int*   dstI   = (const int*)  d_in[3];
  const float* W_node = (const float*)d_in[4];
  const float* b_node = (const float*)d_in[5];
  const float* W_ni   = (const float*)d_in[6];
  const float* W_nj   = (const float*)d_in[7];
  const float* W_fij  = (const float*)d_in[8];
  const float* attn   = (const float*)d_in[9];
  const float* bias   = (const float*)d_in[10];

  const int  N = in_sizes[0] / IN_N;
  const long E = (long)in_sizes[2];

  float* h_out = (float*)d_out;                    // [N, 128]
  float* f_out = (float*)d_out + (size_t)N * HF;   // [E, 128]

  // workspace carve-out (node-level arrays deliberately kept L2-resident)
  char* ws = (char*)d_ws;
  float*    f_ni  = (float*)ws;    ws += (size_t)N * HF * 4;
  float*    f_nj  = (float*)ws;    ws += (size_t)N * HF * 4;
  float*    hproj = (float*)ws;    ws += (size_t)N * HF * 4;
  float*    elog  = (float*)ws;    ws += (size_t)E * 4 * 4;
  unsigned* mkeys = (unsigned*)ws; ws += (size_t)N * 4 * 4;
  float*    sbuf  = (float*)ws;    ws += (size_t)N * 4 * 4;
  _Float16* wpack = (_Float16*)ws;                 // 57344 halves (transposed)

  pack_weights<<<(57344 + 255) / 256, 256, 0, stream>>>(W_ni, W_nj, W_node, W_fij, wpack);

  init_buffers<<<((size_t)N * HF + 255) / 256, 256, 0, stream>>>(h_out, mkeys, sbuf, N);

  dim3 g1((unsigned)((N + 63) / 64), 3);
  node_gemm<<<g1, 256, 0, stream>>>(nfeats, wpack, b_node, f_ni, f_nj, hproj, N);

  edge_gemm<<<(unsigned)((E + 63) / 64), 256, 0, stream>>>(
      efeats, wpack + 49152, f_ni, f_nj, bias, attn, srcI, dstI,
      f_out, elog, mkeys, (int)E);

  softmax_norm<<<(unsigned)((E * 4 + 255) / 256), 256, 0, stream>>>(dstI, mkeys, elog, sbuf, E);

  aggregate<<<(unsigned)((E * 32 + 255) / 256), 256, 0, stream>>>(srcI, dstI, hproj,
                                                                  elog, sbuf, h_out, E);
}
